// MultiHeadAttention_65481071408036
// MI455X (gfx1250) — matrix operand order repaired and checked by reading, not compiled, not run on hardware
//
#include <hip/hip_runtime.h>
#include <hip/hip_bf16.h>
#include <stdint.h>

#define B_ 4
#define T_ 2048
#define C_ 1024
#define H_ 16
#define D_ 64

typedef _Float16 f16;
typedef __attribute__((ext_vector_type(16))) _Float16 v16h;
typedef __attribute__((ext_vector_type(8)))  _Float16 v8h;
typedef __attribute__((ext_vector_type(8)))  float    v8f;

#define WMMA(a, b, c) __builtin_amdgcn_wmma_f32_16x16x32_f16( \
    false, (a), false, (b), (short)0, (c), false, false)

// ---------------------------------------------------------------------------
// CDNA5 async global->LDS copy (ASYNCcnt-tracked, CDNA5 ISA)
typedef __attribute__((address_space(3))) void lds_void_t;

__device__ __forceinline__ uint32_t lds_addr_of(void* p) {
    return (uint32_t)(uintptr_t)(lds_void_t*)p;
}
__device__ __forceinline__ void async_copy16(uint32_t lds_addr, const void* gptr) {
    asm volatile("global_load_async_to_lds_b128 %0, %1, off"
                 :: "v"(lds_addr), "v"((uint64_t)(uintptr_t)gptr)
                 : "memory");
}
__device__ __forceinline__ void async_wait0() {
    asm volatile("s_wait_asynccnt 0x0" ::: "memory");
}

// ---------------------------------------------------------------------------
// Fragment loaders (CDNA5 ISA 7.12.2 layouts, wave32)
__device__ __forceinline__ v16h load_a_frag(const f16* __restrict__ base, int ld, int lane) {
    const int m  = lane & 15;
    const int kh = (lane >> 4) & 1;
    const f16* p = base + (size_t)m * ld + kh * 8;
    v8h lo = *(const v8h*)(p);
    v8h hi = *(const v8h*)(p + 16);
    v16h a;
#pragma unroll
    for (int i = 0; i < 8; ++i) { a[i] = lo[i]; a[i + 8] = hi[i]; }
    return a;
}

__device__ __forceinline__ v16h load_b_fragT(const f16* __restrict__ baseT, int ld, int lane) {
    const int n  = lane & 15;
    const int kh = (lane >> 4) & 1;
    const f16* p = baseT + (size_t)n * ld + kh * 8;
    v8h lo = *(const v8h*)(p);
    v8h hi = *(const v8h*)(p + 16);
    v16h b;
#pragma unroll
    for (int i = 0; i < 8; ++i) { b[i] = lo[i]; b[i + 8] = hi[i]; }
    return b;
}

__device__ __forceinline__ void load_b4(v16h (&b)[4], const f16* __restrict__ wT,
                                        int n0, int k0, int lane) {
#pragma unroll
    for (int t = 0; t < 4; ++t)
        b[t] = load_b_fragT(wT + (size_t)(n0 + t * 16) * C_ + k0, C_, lane);
}

__device__ __forceinline__ void gemm_step(const f16* As, const v16h (&b)[4],
                                          v8f (&acc)[2][4], int lane) {
    v16h a0 = load_a_frag(As,           32, lane);
    v16h a1 = load_a_frag(As + 16 * 32, 32, lane);
#pragma unroll
    for (int t = 0; t < 4; ++t) {
        acc[0][t] = WMMA(a0, b[t], acc[0][t]);
        acc[1][t] = WMMA(a1, b[t], acc[1][t]);
    }
}

// ---------------------------------------------------------------------------
__global__ void __launch_bounds__(256)
cast_f16_kernel(const float* __restrict__ in, f16* __restrict__ out, int n) {
    int i = blockIdx.x * 256 + threadIdx.x;
    if (i < n) out[i] = (f16)in[i];
}

__global__ void __launch_bounds__(256)
transpose_cast_kernel(const float* __restrict__ in, f16* __restrict__ out, int R, int Cc) {
    __shared__ float tile[32][33];
    const int c0 = blockIdx.x * 32;
    const int r0 = blockIdx.y * 32;
    const int tx = threadIdx.x & 31;
    const int ty = threadIdx.x >> 5;
#pragma unroll
    for (int j = 0; j < 32; j += 8)
        tile[ty + j][tx] = in[(size_t)(r0 + ty + j) * Cc + (c0 + tx)];
    __syncthreads();
#pragma unroll
    for (int j = 0; j < 32; j += 8)
        out[(size_t)(c0 + ty + j) * R + (r0 + tx)] = (f16)tile[tx][ty + j];
}

// ---------------------------------------------------------------------------
// QKV projection. Block = 8 waves sharing a 32-row A band (async double-buffered
// LDS, ping-pong). Wave tile 32x64, B-frags ping-pong in registers (no copies).
__global__ void __launch_bounds__(256)
qkv_gemm_kernel(const f16* __restrict__ xh, const f16* __restrict__ wT,
                f16* __restrict__ qh, f16* __restrict__ kbuf, f16* __restrict__ vT) {
    __shared__ __align__(16) f16 At[2][32 * 32];
    const int tid  = threadIdx.x;
    const int lane = tid & 31;
    const int slot = tid >> 5;
    const int m0 = blockIdx.y * 32;
    const int n0 = blockIdx.x * 512 + slot * 64;
    const int arow = tid >> 2, aoff = (tid & 3) * 8;

    auto stageA = [&](int buf, int k0) {
        if (tid < 128)
            async_copy16(lds_addr_of(&At[buf][arow * 32 + aoff]),
                         xh + (size_t)(m0 + arow) * C_ + k0 + aoff);
    };

    v8f acc[2][4] = {};
    v16h b0[4], b1[4];

    stageA(0, 0);
    load_b4(b0, wT, n0, 0, lane);
    async_wait0();
    __syncthreads();

    for (int k0 = 0; k0 < C_; k0 += 64) {
        // even: consume At[0]/b0, prefetch At[1]/b1 (k0+32 < C_ always)
        stageA(1, k0 + 32);
        load_b4(b1, wT, n0, k0 + 32, lane);
        gemm_step(&At[0][0], b0, acc, lane);
        async_wait0();
        __syncthreads();
        // odd: consume At[1]/b1, prefetch At[0]/b0
        if (k0 + 64 < C_) {
            stageA(0, k0 + 64);
            load_b4(b0, wT, n0, k0 + 64, lane);
        }
        gemm_step(&At[1][0], b1, acc, lane);
        async_wait0();
        __syncthreads();
    }

    const int khf = lane >> 4;
    const int nn  = lane & 15;
#pragma unroll
    for (int rt = 0; rt < 2; ++rt)
#pragma unroll
    for (int t = 0; t < 4; ++t) {
        const int n     = n0 + t * 16 + nn;
        const int which = n >> 10;           // 0=q 1=k 2=v
        const int c     = n & (C_ - 1);
        const int h     = c >> 6;
        const int d     = c & 63;
#pragma unroll
        for (int r = 0; r < 8; ++r) {
            const int m  = m0 + rt * 16 + r + 8 * khf;
            const int bb = m >> 11;
            const int tt = m & (T_ - 1);
            const float y = acc[rt][t][r];
            const size_t bhd = (size_t)bb * H_ + h;
            if (which == 0)      qh  [(bhd * T_ + tt) * D_ + d] = (f16)(y * 0.125f);
            else if (which == 1) kbuf[(bhd * T_ + tt) * D_ + d] = (f16)y;
            else                 vT  [(bhd * D_ + d) * T_ + tt] = (f16)y;
        }
    }
}

// ---------------------------------------------------------------------------
// Flash attention. 8 waves per block share one (b,h); K/V 32-key blocks async-
// staged into ping-pong LDS buffers once per block. Wave = 16 query rows.
__global__ void __launch_bounds__(256)
attn_kernel(const f16* __restrict__ qh, const f16* __restrict__ kbuf,
            const f16* __restrict__ vT, f16* __restrict__ oh) {
    __shared__ __align__(16) f16 Kt[2][32 * 64];   // [key-row][d]
    __shared__ __align__(16) f16 Vt[2][64 * 32];   // [d][key-row]
    __shared__ __align__(16) f16 Pt[8][16 * 32];   // per-wave P staging

    const int tid   = threadIdx.x;
    const int lane  = tid & 31;
    const int wslot = tid >> 5;
    const int wave  = blockIdx.x * 8 + wslot;
    const int bh    = wave >> 7;
    const int qt    = wave & 127;
    const int bb    = bh >> 4;
    const int h     = bh & 15;

    const f16* Q  = qh   + (size_t)bh * T_ * D_;
    const f16* Kg = kbuf + (size_t)bh * T_ * D_;
    const f16* Vg = vT   + (size_t)bh * D_ * T_;

    const int vr = tid >> 2, vo = (tid & 3) * 8;

    auto stageKV = [&](int buf, int kb) {
        async_copy16(lds_addr_of(&Kt[buf][tid * 8]), Kg + (size_t)kb * D_ + tid * 8);
        async_copy16(lds_addr_of(&Vt[buf][vr * 32 + vo]), Vg + (size_t)vr * T_ + kb + vo);
    };

    const int q0 = qt * 16;
    const v16h qa0 = load_a_frag(Q + (size_t)q0 * D_,      D_, lane);
    const v16h qa1 = load_a_frag(Q + (size_t)q0 * D_ + 32, D_, lane);

    v8f o[4] = {};
    float mrun[8], lrun[8];
#pragma unroll
    for (int r = 0; r < 8; ++r) { mrun[r] = -1e30f; lrun[r] = 0.0f; }

    const int nn  = lane & 15;
    const int khf = lane >> 4;
    f16* myp = &Pt[wslot][0];

    auto attn_step = [&](const f16* Kb, const f16* Vb) {
        v8f s0 = {}, s1 = {};
        {
            v16h c00 = load_b_fragT(Kb,                64, lane);
            v16h c01 = load_b_fragT(Kb + 32,           64, lane);
            v16h c10 = load_b_fragT(Kb + 16 * 64,      64, lane);
            v16h c11 = load_b_fragT(Kb + 16 * 64 + 32, 64, lane);
            s0 = WMMA(qa0, c00, s0); s0 = WMMA(qa1, c01, s0);
            s1 = WMMA(qa0, c10, s1); s1 = WMMA(qa1, c11, s1);
        }
        float al[8];
#pragma unroll
        for (int r = 0; r < 8; ++r) {
            float mx = fmaxf(s0[r], s1[r]);
            mx = fmaxf(mx, __shfl_xor(mx, 1, 32));
            mx = fmaxf(mx, __shfl_xor(mx, 2, 32));
            mx = fmaxf(mx, __shfl_xor(mx, 4, 32));
            mx = fmaxf(mx, __shfl_xor(mx, 8, 32));
            const float mnew = fmaxf(mrun[r], mx);
            const float a_   = __expf(mrun[r] - mnew);
            const float p0   = __expf(s0[r] - mnew);
            const float p1   = __expf(s1[r] - mnew);
            float rs = p0 + p1;
            rs += __shfl_xor(rs, 1, 32);
            rs += __shfl_xor(rs, 2, 32);
            rs += __shfl_xor(rs, 4, 32);
            rs += __shfl_xor(rs, 8, 32);
            lrun[r] = lrun[r] * a_ + rs;
            mrun[r] = mnew;
            al[r]   = a_;
            const int m = r + 8 * khf;
            myp[m * 32 + nn]      = (f16)p0;
            myp[m * 32 + 16 + nn] = (f16)p1;
        }
#pragma unroll
        for (int t = 0; t < 4; ++t)
#pragma unroll
            for (int r = 0; r < 8; ++r) o[t][r] *= al[r];
        const v16h pa = load_a_frag(myp, 32, lane);
#pragma unroll
        for (int t = 0; t < 4; ++t) {
            v16h vb = load_b_fragT(Vb + t * 16 * 32, 32, lane);
            o[t] = WMMA(pa, vb, o[t]);
        }
    };

    stageKV(0, 0);
    async_wait0();
    __syncthreads();

    for (int kb = 0; kb < T_; kb += 64) {
        // even: consume buf0, prefetch buf1 at kb+32 (always valid)
        stageKV(1, kb + 32);
        attn_step(&Kt[0][0], &Vt[0][0]);
        async_wait0();
        __syncthreads();
        // odd: consume buf1, prefetch buf0 at kb+64
        if (kb + 64 < T_) stageKV(0, kb + 64);
        attn_step(&Kt[1][0], &Vt[1][0]);
        async_wait0();
        __syncthreads();
    }

#pragma unroll
    for (int t = 0; t < 4; ++t)
#pragma unroll
        for (int r = 0; r < 8; ++r) {
            const int m = q0 + r + 8 * khf;
            const int d = t * 16 + nn;
            oh[((size_t)bb * T_ + m) * C_ + h * D_ + d] = (f16)(o[t][r] / lrun[r]);
        }
}

// ---------------------------------------------------------------------------
// Output projection: out = oh @ w_out + b_out (f32). Same pipelined blocking.
__global__ void __launch_bounds__(256)
out_gemm_kernel(const f16* __restrict__ ohb, const f16* __restrict__ wT,
                const float* __restrict__ bias, float* __restrict__ out) {
    __shared__ __align__(16) f16 At[2][32 * 32];
    const int tid  = threadIdx.x;
    const int lane = tid & 31;
    const int slot = tid >> 5;
    const int m0 = blockIdx.y * 32;
    const int n0 = blockIdx.x * 512 + slot * 64;
    const int arow = tid >> 2, aoff = (tid & 3) * 8;

    auto stageA = [&](int buf, int k0) {
        if (tid < 128)
            async_copy16(lds_addr_of(&At[buf][arow * 32 + aoff]),
                         ohb + (size_t)(m0 + arow) * C_ + k0 + aoff);
    };

    v8f acc[2][4] = {};
    v16h b0[4], b1[4];

    stageA(0, 0);
    load_b4(b0, wT, n0, 0, lane);
    async_wait0();
    __syncthreads();

    for (int k0 = 0; k0 < C_; k0 += 64) {
        stageA(1, k0 + 32);
        load_b4(b1, wT, n0, k0 + 32, lane);
        gemm_step(&At[0][0], b0, acc, lane);
        async_wait0();
        __syncthreads();
        if (k0 + 64 < C_) {
            stageA(0, k0 + 64);
            load_b4(b0, wT, n0, k0 + 64, lane);
        }
        gemm_step(&At[1][0], b1, acc, lane);
        async_wait0();
        __syncthreads();
    }

    const int khf = lane >> 4;
    const int nn  = lane & 15;
#pragma unroll
    for (int rt = 0; rt < 2; ++rt)
#pragma unroll
    for (int t = 0; t < 4; ++t) {
        const int n    = n0 + t * 16 + nn;
        const float bv = bias[n];
#pragma unroll
        for (int r = 0; r < 8; ++r) {
            const int m = m0 + rt * 16 + r + 8 * khf;
            out[(size_t)m * C_ + n] = acc[rt][t][r] + bv;
        }
    }
}

// ---------------------------------------------------------------------------
extern "C" void kernel_launch(void* const* d_in, const int* in_sizes, int n_in,
                              void* d_out, int out_size, void* d_ws, size_t ws_size,
                              hipStream_t stream) {
    (void)in_sizes; (void)n_in; (void)out_size; (void)ws_size;
    const float* x     = (const float*)d_in[0];
    const float* w_qkv = (const float*)d_in[1];
    const float* w_out = (const float*)d_in[2];
    const float* b_out = (const float*)d_in[3];
    float* out = (float*)d_out;

    char* ws = (char*)d_ws;
    const size_t SZ_XH    = (size_t)B_ * T_ * C_ * 2;
    const size_t SZ_WQKVT = (size_t)3 * C_ * C_ * 2;
    const size_t SZ_WOUTT = (size_t)C_ * C_ * 2;
    const size_t SZ_HEAD  = (size_t)B_ * H_ * T_ * D_ * 2;
    f16* xh    = (f16*)(ws);
    f16* wqkvT = (f16*)(ws + SZ_XH);
    f16* woutT = (f16*)(ws + SZ_XH + SZ_WQKVT);
    f16* qh    = (f16*)(ws + SZ_XH + SZ_WQKVT + SZ_WOUTT);
    f16* kbuf  = (f16*)(ws + SZ_XH + SZ_WQKVT + SZ_WOUTT + SZ_HEAD);
    f16* vT    = (f16*)(ws + SZ_XH + SZ_WQKVT + SZ_WOUTT + 2 * SZ_HEAD);
    f16* oh    = (f16*)(ws + SZ_XH + SZ_WQKVT + SZ_WOUTT + 3 * SZ_HEAD);

    const int nx = B_ * T_ * C_;
    cast_f16_kernel<<<(nx + 255) / 256, 256, 0, stream>>>(x, xh, nx);
    transpose_cast_kernel<<<dim3((3 * C_) / 32, C_ / 32), 256, 0, stream>>>(w_qkv, wqkvT, C_, 3 * C_);
    transpose_cast_kernel<<<dim3(C_ / 32, C_ / 32), 256, 0, stream>>>(w_out, woutT, C_, C_);

    qkv_gemm_kernel<<<dim3(6, 256), 256, 0, stream>>>(xh, wqkvT, qh, kbuf, vT);
    attn_kernel<<<1024, 256, 0, stream>>>(qh, kbuf, vT, oh);
    out_gemm_kernel<<<dim3(2, 256), 256, 0, stream>>>(oh, woutT, b_out, out);
}
